// GraphAttentionLayer_84954453115175
// MI455X (gfx1250) — matrix operand order repaired and checked by reading, not compiled, not run on hardware
//
#include <hip/hip_runtime.h>

typedef _Float16 f16;
typedef __attribute__((ext_vector_type(16))) _Float16 v16h;
typedef __attribute__((ext_vector_type(8)))  _Float16 v8h;
typedef __attribute__((ext_vector_type(8)))  float    v8f;

#define BB   8
#define NN   1024
#define DDIM 512
#define HD   64
#define HH   8
#define HB   64      /* H*B */
#define ROWS 8192    /* B*N */
#define NEG_BIG (-3.0e38f)
#define BSTR 72      /* padded LDS row stride (halves): 144B -> conflict-free, 16B-aligned */

// ---------------------------------------------------------------- CDNA5 async global->LDS
__device__ __forceinline__ void async_b128(unsigned lds_byte_off, const f16* gaddr) {
  asm volatile("global_load_async_to_lds_b128 %0, %1, off"
               :: "v"(lds_byte_off), "v"(gaddr) : "memory");
}
__device__ __forceinline__ void wait_async0() {
  asm volatile("s_wait_asynccnt 0x0" ::: "memory");
}
__device__ __forceinline__ unsigned lds_off(const void* p) {
  return (unsigned)(size_t)p;   // generic LDS ptr = aperture_hi | lds_offset -> low 32 bits
}

// ---------------------------------------------------------------- helpers
__global__ void cvt_f32_f16(const float* __restrict__ in, f16* __restrict__ out, int n) {
  int i = blockIdx.x * blockDim.x + threadIdx.x;
  if (i < n) out[i] = (f16)in[i];
}

// Wt[c][k] = W[k][c]  (512x512), f32 -> f16 transposed so GEMM B-fragments are contiguous
__global__ void trans_w(const float* __restrict__ W, f16* __restrict__ Wt) {
  int i = blockIdx.x * blockDim.x + threadIdx.x;   // 512*512 threads
  int k = i >> 9, c = i & 511;
  Wt[(size_t)c * DDIM + k] = (f16)W[i];
}

// ---------------------------------------------------------------- GEMM
// C[8192,512] = A[8192,512] @ B[512,512], B given transposed (Bt[n][k]).
// One wave -> 16x64 output tile; 8 waves / block share one 64-col B panel.
// B panel is staged to LDS via async global->LDS b128, double-buffered in
// 64-deep K chunks (ASYNCcnt pipeline); WMMA consumes from LDS.
// mode 0: out f16 head-major [H*B][N][64], val=(acc+bias)*scale   (QKV)
// mode 1: out f16 row-major, relu(acc+bias)                       (FFN1)
// mode 2: out f32 row-major, acc+bias                             (FFN2)
__global__ void gemm_kernel(const f16* __restrict__ A, const f16* __restrict__ Bt,
                            const float* __restrict__ bias, void* __restrict__ out,
                            int mode, float scale) {
  __shared__ f16 Bs[2][64 * BSTR];           // 2 x 9216 B double buffer
  int tid  = threadIdx.x;
  int wave = tid >> 5;
  int lane = tid & 31;
  int wt   = blockIdx.x * 8 + wave;
  int tm   = wt & 511;            // 512 M-tiles of 16
  int tn   = wt >> 9;             // 8 N-tiles of 64 (uniform across the block)
  int m0 = tm * 16, n0 = tn * 64;

  int lrow = lane & 15;
  int hi   = lane >> 4;           // 0 or 1
  int ka0  = hi * 16;             // A frag: lanes<16 -> K 0..15 ; lanes>=16 -> K 16..31
  int ka1  = hi * 16 + 8;
  int kb0  = hi * 16;             // B frag: lanes<16 -> K 0..15 ; lanes>=16 -> K 16..31

  const f16* Arow = A + (size_t)(m0 + lrow) * DDIM;

  // async staging: thread t handles col c = t>>2, 32 bytes at k-offset (t&3)*16 halves
  int cc = tid >> 2;
  int ko = (tid & 3) * 16;
  const f16* gsrc = Bt + (size_t)(n0 + cc) * DDIM + ko;     // + ck*64 per chunk
  unsigned lb0 = lds_off(&Bs[0][cc * BSTR + ko]);
  unsigned lb1 = lds_off(&Bs[1][cc * BSTR + ko]);

  // preload chunk 0 into buffer 0
  async_b128(lb0,      gsrc);
  async_b128(lb0 + 16, gsrc + 8);
  wait_async0();
  __syncthreads();

  v8f acc0 = {}, acc1 = {}, acc2 = {}, acc3 = {};

  for (int ck = 0; ck < DDIM / 64; ++ck) {
    if (ck < DDIM / 64 - 1) {                 // prefetch next chunk into the other buffer
      const f16* gn = gsrc + (ck + 1) * 64;
      unsigned lb = (ck & 1) ? lb0 : lb1;
      async_b128(lb,      gn);
      async_b128(lb + 16, gn + 8);
    }
    const f16* Bbuf = &Bs[ck & 1][0];
#pragma unroll
    for (int s2 = 0; s2 < 2; ++s2) {          // two 32-deep K steps per chunk
      int k  = ck * 64 + s2 * 32;
      int kl = s2 * 32;
      union { v16h v; v8h h[2]; } ua, ub0, ub1, ub2, ub3;
      ua.h[0] = *(const v8h*)(Arow + k + ka0);
      ua.h[1] = *(const v8h*)(Arow + k + ka1);
      const f16* bp0 = Bbuf + ( 0 + lrow) * BSTR + kl + kb0;
      const f16* bp1 = Bbuf + (16 + lrow) * BSTR + kl + kb0;
      const f16* bp2 = Bbuf + (32 + lrow) * BSTR + kl + kb0;
      const f16* bp3 = Bbuf + (48 + lrow) * BSTR + kl + kb0;
      ub0.h[0] = *(const v8h*)(bp0); ub0.h[1] = *(const v8h*)(bp0 + 8);
      ub1.h[0] = *(const v8h*)(bp1); ub1.h[1] = *(const v8h*)(bp1 + 8);
      ub2.h[0] = *(const v8h*)(bp2); ub2.h[1] = *(const v8h*)(bp2 + 8);
      ub3.h[0] = *(const v8h*)(bp3); ub3.h[1] = *(const v8h*)(bp3 + 8);
      acc0 = __builtin_amdgcn_wmma_f32_16x16x32_f16(false, ua.v, false, ub0.v, (short)0, acc0, false, false);
      acc1 = __builtin_amdgcn_wmma_f32_16x16x32_f16(false, ua.v, false, ub1.v, (short)0, acc1, false, false);
      acc2 = __builtin_amdgcn_wmma_f32_16x16x32_f16(false, ua.v, false, ub2.v, (short)0, acc2, false, false);
      acc3 = __builtin_amdgcn_wmma_f32_16x16x32_f16(false, ua.v, false, ub3.v, (short)0, acc3, false, false);
    }
    wait_async0();                            // next buffer fully in LDS
    __syncthreads();                          // ... for every wave in the block
  }

  // C layout: VGPR g -> row m0+g (lanes 0-15) / m0+8+g (lanes 16-31), col = n0+sub*16+lrow
  int rbase = m0 + hi * 8;
  v8f accs[4] = {acc0, acc1, acc2, acc3};
#pragma unroll
  for (int s = 0; s < 4; ++s) {
    int col = n0 + s * 16 + lrow;
    float bv = bias[col];
#pragma unroll
    for (int g = 0; g < 8; ++g) {
      int   r = rbase + g;
      float v = accs[s][g] + bv;
      if (mode == 0) {
        v *= scale;
        int h = col >> 6, d = col & 63, b = r >> 10, n = r & 1023;
        ((f16*)out)[((((size_t)h * BB + b) * NN + n) << 6) + d] = (f16)v;
      } else if (mode == 1) {
        v = v > 0.f ? v : 0.f;
        ((f16*)out)[(size_t)r * DDIM + col] = (f16)v;
      } else {
        ((float*)out)[(size_t)r * DDIM + col] = v;
      }
    }
  }
}

// ---------------------------------------------------------------- fused attention
// One block (4 waves) = one (head*batch hb, 16-query tile).
// Phase 1: S[16][1024] scores via WMMA into LDS (waves split key tiles).
// Phase 2: per row -> iterative top-32 (one survivor per lane), softmax, gather V.
__global__ void attn_kernel(const f16* __restrict__ Q, const f16* __restrict__ K,
                            const f16* __restrict__ V, float* __restrict__ out) {
  extern __shared__ float S[];                 // 16*1024 f32 = 64 KB
  int hb = blockIdx.x >> 6;
  int q0 = (blockIdx.x & 63) * 16;
  int wave = threadIdx.x >> 5;
  int lane = threadIdx.x & 31;
  int lrow = lane & 15;
  int hi   = lane >> 4;

  const f16* Qh = Q + (size_t)hb * NN * HD;
  const f16* Kh = K + (size_t)hb * NN * HD;
  const f16* Vh = V + (size_t)hb * NN * HD;

  // A fragments for the 16 query rows (K-dim = 64 -> two 32-chunks)
  int ka0 = hi * 16, ka1 = hi * 16 + 8, kb0 = hi * 16;
  const f16* Ar = Qh + (size_t)(q0 + lrow) * HD;
  union { v16h v; v8h h[2]; } u0, u1;
  u0.h[0] = *(const v8h*)(Ar + 0  + ka0);
  u0.h[1] = *(const v8h*)(Ar + 0  + ka1);
  u1.h[0] = *(const v8h*)(Ar + 32 + ka0);
  u1.h[1] = *(const v8h*)(Ar + 32 + ka1);
  v16h a0 = u0.v, a1 = u1.v;

  for (int kt = wave; kt < NN / 16; kt += 4) {
    const f16* Bc = Kh + (size_t)(kt * 16 + lrow) * HD + kb0;
    v16h b0 = *(const v16h*)(Bc);
    v16h b1 = *(const v16h*)(Bc + 32);
    v8f acc = {};
    acc = __builtin_amdgcn_wmma_f32_16x16x32_f16(false, a0, false, b0, (short)0, acc, false, false);
    acc = __builtin_amdgcn_wmma_f32_16x16x32_f16(false, a1, false, b1, (short)0, acc, false, false);
    int col = kt * 16 + lrow;
#pragma unroll
    for (int g = 0; g < 8; ++g)
      S[(size_t)(hi * 8 + g) * NN + col] = acc[g];
  }
  __syncthreads();

  // Phase 2: 4 rows per wave
  for (int rr = 0; rr < 4; ++rr) {
    int r = wave * 4 + rr;
    float val[32];
#pragma unroll
    for (int j = 0; j < 32; ++j) val[j] = S[(size_t)r * NN + lane + 32 * j];

    unsigned used = 0u;
    float selV = NEG_BIG;
    int   selI = 0;
    for (int t = 0; t < 32; ++t) {
      // local argmax over unused entries (smallest j wins ties -> unique global winner)
      float m = NEG_BIG; int jm = 0;
#pragma unroll
      for (int j = 0; j < 32; ++j) {
        bool take = (((used >> j) & 1u) == 0u) && (val[j] > m);
        if (take) { m = val[j]; jm = j; }
      }
      int gi = lane + (jm << 5);
      // cross-lane argmax (value desc, index asc on ties)
#pragma unroll
      for (int o = 16; o > 0; o >>= 1) {
        float mo = __shfl_xor(m, o);
        int   io = __shfl_xor(gi, o);
        if (mo > m || (mo == m && io < gi)) { m = mo; gi = io; }
      }
      if (lane == t) { selV = m; selI = gi; }
      if ((gi & 31) == lane) used |= 1u << (gi >> 5);
    }

    // softmax over the 32 survivors (selection is descending -> lane 0 holds the max)
    float mx = __shfl(selV, 0);
    float e  = __expf(selV - mx);
    float sum = e;
#pragma unroll
    for (int o = 16; o > 0; o >>= 1) sum += __shfl_xor(sum, o);
    float wgt = e / sum;

    // attn @ V : each lane accumulates dims (lane, lane+32)
    float acc0 = 0.f, acc1 = 0.f;
    for (int t = 0; t < 32; ++t) {
      float wt = __shfl(wgt, t);
      int   it = __shfl(selI, t);
      const f16* vp = Vh + (size_t)it * HD;
      acc0 += wt * (float)vp[lane];
      acc1 += wt * (float)vp[lane + 32];
    }
    int h = hb >> 3, b = hb & 7, q = q0 + r;
    float* op = out + ((size_t)(b * NN + q)) * DDIM + h * HD;
    op[lane]      = acc0;
    op[lane + 32] = acc1;
  }
}

// ---------------------------------------------------------------- residual + LayerNorm
// y = LN(A + Bm) * g + be ; optionally also emit f16 copy for the next GEMM.
__global__ void ln_kernel(const float* __restrict__ A, const float* __restrict__ Bm,
                          const float* __restrict__ g, const float* __restrict__ be,
                          float* __restrict__ out, f16* __restrict__ out16) {
  int row = blockIdx.x;
  int tid = threadIdx.x;                  // 256 threads, 2 elems each
  const float* ap = A  + (size_t)row * DDIM;
  const float* bp = Bm + (size_t)row * DDIM;
  float v0 = ap[tid] + bp[tid];
  float v1 = ap[tid + 256] + bp[tid + 256];
  float s  = v0 + v1;
  float s2 = v0 * v0 + v1 * v1;
#pragma unroll
  for (int o = 16; o > 0; o >>= 1) { s += __shfl_xor(s, o); s2 += __shfl_xor(s2, o); }
  __shared__ float rs[8], rs2[8];
  int wave = tid >> 5, lane = tid & 31;
  if (lane == 0) { rs[wave] = s; rs2[wave] = s2; }
  __syncthreads();
  if (wave == 0) {
    s  = (lane < 8) ? rs[lane]  : 0.f;
    s2 = (lane < 8) ? rs2[lane] : 0.f;
#pragma unroll
    for (int o = 4; o > 0; o >>= 1) { s += __shfl_xor(s, o); s2 += __shfl_xor(s2, o); }
    if (lane == 0) { rs[0] = s; rs2[0] = s2; }
  }
  __syncthreads();
  s = rs[0]; s2 = rs2[0];
  float mean = s * (1.f / DDIM);
  float var  = s2 * (1.f / DDIM) - mean * mean;
  float rstd = rsqrtf(var + 1e-5f);
  float y0 = (v0 - mean) * rstd * g[tid]       + be[tid];
  float y1 = (v1 - mean) * rstd * g[tid + 256] + be[tid + 256];
  out[(size_t)row * DDIM + tid]       = y0;
  out[(size_t)row * DDIM + tid + 256] = y1;
  if (out16) {
    out16[(size_t)row * DDIM + tid]       = (f16)y0;
    out16[(size_t)row * DDIM + tid + 256] = (f16)y1;
  }
}

// ---------------------------------------------------------------- launcher
extern "C" void kernel_launch(void* const* d_in, const int* in_sizes, int n_in,
                              void* d_out, int out_size, void* d_ws, size_t ws_size,
                              hipStream_t stream) {
  (void)in_sizes; (void)n_in; (void)out_size; (void)ws_size;
  const float* src = (const float*)d_in[0];
  const float* tgt = (const float*)d_in[1];
  const float* Wq  = (const float*)d_in[2];
  const float* bq  = (const float*)d_in[3];
  const float* Wk  = (const float*)d_in[4];
  const float* bk  = (const float*)d_in[5];
  const float* Wv  = (const float*)d_in[6];
  const float* bv  = (const float*)d_in[7];
  const float* W1  = (const float*)d_in[8];
  const float* b1  = (const float*)d_in[9];
  const float* W2  = (const float*)d_in[10];
  const float* b2  = (const float*)d_in[11];
  const float* g1  = (const float*)d_in[12];
  const float* be1 = (const float*)d_in[13];
  const float* g2  = (const float*)d_in[14];
  const float* be2 = (const float*)d_in[15];
  float* out = (float*)d_out;

  const size_t SZ_ACT16 = (size_t)ROWS * DDIM * 2;   // 8 MB
  const size_t SZ_W16   = (size_t)DDIM * DDIM * 2;   // 0.5 MB
  const size_t SZ_ACT32 = (size_t)ROWS * DDIM * 4;   // 16 MB

  char* w = (char*)d_ws;
  size_t off = 0;
  f16* h_src = (f16*)(w + off); off += SZ_ACT16;
  f16* h_tgt = (f16*)(w + off); off += SZ_ACT16;
  f16* Wqt = (f16*)(w + off); off += SZ_W16;
  f16* Wkt = (f16*)(w + off); off += SZ_W16;
  f16* Wvt = (f16*)(w + off); off += SZ_W16;
  f16* W1t = (f16*)(w + off); off += SZ_W16;
  f16* W2t = (f16*)(w + off); off += SZ_W16;
  f16* q16 = (f16*)(w + off); off += SZ_ACT16;      // [H*B][N][64]
  f16* k16 = (f16*)(w + off); off += SZ_ACT16;
  f16* v16 = (f16*)(w + off); off += SZ_ACT16;
  float* attn_out = (float*)(w + off); off += SZ_ACT32;
  float* x   = (float*)(w + off); off += SZ_ACT32;  // post-LN1, f32
  f16*  xh   = (f16*)(w + off); off += SZ_ACT16;    // post-LN1, f16
  float* ff  = attn_out;                            // alias: attn_out dead after LN1
  f16*  y16  = h_src;                               // alias: h_src dead after K/V GEMMs

  const int NACT = ROWS * DDIM;

  cvt_f32_f16<<<NACT / 256, 256, 0, stream>>>(src, h_src, NACT);
  cvt_f32_f16<<<NACT / 256, 256, 0, stream>>>(tgt, h_tgt, NACT);
  trans_w<<<(DDIM * DDIM) / 256, 256, 0, stream>>>(Wq, Wqt);
  trans_w<<<(DDIM * DDIM) / 256, 256, 0, stream>>>(Wk, Wkt);
  trans_w<<<(DDIM * DDIM) / 256, 256, 0, stream>>>(Wv, Wvt);
  trans_w<<<(DDIM * DDIM) / 256, 256, 0, stream>>>(W1, W1t);
  trans_w<<<(DDIM * DDIM) / 256, 256, 0, stream>>>(W2, W2t);

  // Q scaled by 1/sqrt(D) as in reference
  gemm_kernel<<<512, 256, 0, stream>>>(h_tgt, Wqt, bq, q16, 0, 0.04419417382f);
  gemm_kernel<<<512, 256, 0, stream>>>(h_src, Wkt, bk, k16, 0, 1.0f);
  gemm_kernel<<<512, 256, 0, stream>>>(h_src, Wvt, bv, v16, 0, 1.0f);

  attn_kernel<<<HB * (NN / 16), 128, 16 * NN * sizeof(float), stream>>>(q16, k16, v16, attn_out);

  ln_kernel<<<ROWS, 256, 0, stream>>>(tgt, attn_out, g1, be1, x, xh);

  gemm_kernel<<<512, 256, 0, stream>>>(xh, W1t, b1, y16, 1, 1.0f);   // relu(x@W1+b1) f16
  gemm_kernel<<<512, 256, 0, stream>>>(y16, W2t, b2, ff, 2, 1.0f);   // @W2+b2 f32

  ln_kernel<<<ROWS, 256, 0, stream>>>(x, ff, g2, be2, out, nullptr);
}
